// mLSTMBackendTriton_53678501265638
// MI455X (gfx1250) — matrix operand order repaired and checked by reading, not compiled, not run on hardware
//
#include <hip/hip_runtime.h>

typedef __bf16 bf16;
typedef __bf16 v16bf __attribute__((ext_vector_type(16)));
typedef __bf16 v8bf  __attribute__((ext_vector_type(8)));
typedef float  v8f   __attribute__((ext_vector_type(8)));

#define LOG2E 1.4426950408889634f
#define BTC 64   // chunk length
#define KD  128  // key dim
#define VD  128  // value dim
#define NT  64   // chunks
#define TT  4096 // sequence

#if defined(__has_builtin)
# if __has_builtin(__builtin_amdgcn_global_load_async_to_lds_b32) && __has_builtin(__builtin_amdgcn_s_wait_asynccnt)
#  define USE_ASYNC_GATES 1
# endif
#endif
#ifndef USE_ASYNC_GATES
# define USE_ASYNC_GATES 0
#endif

// ---- WMMA fragment loaders (CDNA5 wave32 bf16 16x16x32) ----
// A (16x32 MxK) from row-major [M][K]: lanes 0-15 row=lane K={0..15}; lanes 16-31 row=lane-16 K={16..31}
__device__ __forceinline__ v16bf frag_a(const bf16* __restrict__ s, int ld, int row0, int k0, int lane) {
  const int hl = lane >> 4, m = lane & 15;
  const bf16* p = s + (row0 + m) * ld + k0 + hl * 16;
  v16bf a;
#pragma unroll
  for (int r = 0; r < 8; ++r) {
    const int kb = 2 * r;
    a[2 * r]     = p[kb];
    a[2 * r + 1] = p[kb + 1];
  }
  return a;
}
// B (32x16 KxN) from N-major source: element (kk,n) at s[(n0+n)*ld + k0+kk]  (16 contiguous bf16 per lane)
__device__ __forceinline__ v16bf frag_bt(const bf16* __restrict__ s, int ld, int k0, int n0, int lane) {
  const int hl = lane >> 4, n = lane & 15;
  const bf16* p = s + (n0 + n) * ld + k0 + hl * 16;
  v16bf b;
#pragma unroll
  for (int r = 0; r < 16; ++r) b[r] = p[r];
  return b;
}

__global__ __launch_bounds__(256)
void mlstm_chunk_kernel(const float* __restrict__ q, const float* __restrict__ k,
                        const float* __restrict__ v, const float* __restrict__ ig,
                        const float* __restrict__ fg, float* __restrict__ out) {
  __shared__ bf16 qs[BTC * KD];    // qbar (row-scaled), row-major [i][K]  (A-operand)
  __shared__ bf16 kb[BTC * KD];    // kbar row-major [j][K]                (B-operand for S, via frag_bt)
  __shared__ bf16 kbT[KD * BTC];   // kbar transposed [K][j]               (A-operand for C update)
  __shared__ bf16 vsT[VD * BTC];   // v transposed [v][j]                  (B-operand for S@V and C update)
  __shared__ bf16 CbT[VD * KD];    // entering state C^T [v][k]            (B-operand for qbar@C)
  __shared__ bf16 Sb[BTC * BTC];   // masked scaled S, row-major [i][j]    (A-operand)
  __shared__ float sB[BTC], sI[BTC], sU[BTC], sA[BTC], sMC[BTC];
  __shared__ float sQf[BTC], sKf[BTC], sRed[BTC], sRn[BTC];
  __shared__ float sN[KD];
  __shared__ float sM[1];
#if USE_ASYNC_GATES
  __shared__ float sFg[2][BTC], sIg[2][BTC];
#endif

  const int tid = threadIdx.x;
  const int wv = tid >> 5, lane = tid & 31;
  const int hl = lane >> 4, lx = lane & 15;
  const int bh = blockIdx.x;
  const float scale = 0.08838834764831845f; // 128^-0.5

  const float* qp = q + (size_t)bh * TT * KD;
  const float* kp = k + (size_t)bh * TT * KD;
  const float* vp = v + (size_t)bh * TT * VD;
  const float* ip = ig + (size_t)bh * TT;
  const float* fp = fg + (size_t)bh * TT;
  float* op = out + (size_t)bh * TT * VD;

  // C state in registers: wave wv owns K-rows [16wv, 16wv+16), 8 V-tiles of 16
  v8f cfr[8];
  v8f vzero = {};
#pragma unroll
  for (int t = 0; t < 8; ++t) cfr[t] = vzero;
  if (tid < KD) sN[tid] = 0.0f;
  if (tid == 0) sM[0] = 0.0f;
#if USE_ASYNC_GATES
  if (tid < BTC) { // async-prefetch gates for chunk 0 (ASYNCcnt path)
    __builtin_amdgcn_global_load_async_to_lds_b32((int*)(fp + tid), (int*)&sFg[0][tid], 0, 0);
    __builtin_amdgcn_global_load_async_to_lds_b32((int*)(ip + tid), (int*)&sIg[0][tid], 0, 0);
  }
#endif
  __syncthreads();

  for (int c = 0; c < NT; ++c) {
    const float m_old = sM[0];
    // ---------- gates: log-sigmoid, to log2 space ----------
#if USE_ASYNC_GATES
    __builtin_amdgcn_s_wait_asynccnt(0);
    if (tid < BTC) {
      const float f = sFg[c & 1][tid];
      const float ls = fminf(f, 0.0f) - log1pf(expf(-fabsf(f)));
      sB[tid] = ls * LOG2E;
      sI[tid] = sIg[c & 1][tid] * LOG2E;
    }
#else
    if (tid < BTC) {
      const float f = fp[c * BTC + tid];
      const float ls = fminf(f, 0.0f) - log1pf(expf(-fabsf(f)));
      sB[tid] = ls * LOG2E;
      sI[tid] = ip[c * BTC + tid] * LOG2E;
    }
#endif
    __syncthreads();
    // inclusive cumsum of f-gates (vecB)
    for (int off = 1; off < BTC; off <<= 1) {
      float x = 0.0f;
      if (tid < BTC) { x = sB[tid]; if (tid >= off) x += sB[tid - off]; }
      __syncthreads();
      if (tid < BTC) sB[tid] = x;
      __syncthreads();
    }
    if (tid < BTC) sU[tid] = sI[tid] - sB[tid];
    __syncthreads();
    // inclusive prefix-max of u[j]=I[j]-B[j]  ->  m_intra[i] = B[i] + premax(u)[i]
    for (int off = 1; off < BTC; off <<= 1) {
      float x = 0.0f;
      if (tid < BTC) { x = sU[tid]; if (tid >= off) x = fmaxf(x, sU[tid - off]); }
      __syncthreads();
      if (tid < BTC) sU[tid] = x;
      __syncthreads();
    }
    const float g = sB[BTC - 1];
    if (tid < BTC) {
      const float a = sI[tid] + g - sB[tid];              // vecA
      sA[tid] = a;
      sRed[tid] = a;
      const float m_intra = sB[tid] + sU[tid];
      const float mc = fmaxf(m_old + sB[tid], m_intra);   // m_comb
      sMC[tid] = mc;
      sQf[tid] = scale * exp2f(sB[tid] + m_old - mc);     // qbar row factor
    }
    __syncthreads();
    for (int off = BTC / 2; off > 0; off >>= 1) {
      if (tid < off) sRed[tid] = fmaxf(sRed[tid], sRed[tid + off]);
      __syncthreads();
    }
    const float m_next = fmaxf(g + m_old, sRed[0]);
    const float dec  = exp2f(g + m_old - m_next);
    const float rdec = exp2f(m_next - g - m_old);         // S = (qbar@kbar^T)*rdec
    if (tid < BTC) sKf[tid] = exp2f(sA[tid] - m_next);    // kbar row factor
#if USE_ASYNC_GATES
    if (c + 1 < NT && tid < BTC) { // async-prefetch next chunk's gates
      __builtin_amdgcn_global_load_async_to_lds_b32(
          (int*)(fp + (c + 1) * BTC + tid), (int*)&sFg[(c + 1) & 1][tid], 0, 0);
      __builtin_amdgcn_global_load_async_to_lds_b32(
          (int*)(ip + (c + 1) * BTC + tid), (int*)&sIg[(c + 1) & 1][tid], 0, 0);
    }
#endif
    __syncthreads();

    // ---------- load + convert chunk ----------
    const size_t cb = (size_t)c * BTC * KD;
    // q: row-major, pre-scaled by sQf
    const float4* q4 = (const float4*)(qp + cb);
#pragma unroll 2
    for (int i4 = tid; i4 < BTC * KD / 4; i4 += 256) {
      const int row = i4 >> 5, c0 = (i4 & 31) * 4;
      const float qf = sQf[row];
      const float4 a = q4[i4];
      bf16* d = qs + row * KD + c0;
      d[0] = (bf16)(a.x * qf); d[1] = (bf16)(a.y * qf);
      d[2] = (bf16)(a.z * qf); d[3] = (bf16)(a.w * qf);
    }
    // k, v: 4x4 blocks -> kbar row-major + transposed, v transposed (contiguous b64 LDS stores)
#pragma unroll 2
    for (int blk = tid; blk < (BTC / 4) * (KD / 4); blk += 256) {
      const int r0 = (blk >> 5) * 4;
      const int c0 = (blk & 31) * 4;
      float kr[4][4], vr[4][4];
#pragma unroll
      for (int i = 0; i < 4; ++i) {
        const float kf = sKf[r0 + i];
        const float4 t = *(const float4*)(kp + cb + (size_t)(r0 + i) * KD + c0);
        kr[i][0] = t.x * kf; kr[i][1] = t.y * kf; kr[i][2] = t.z * kf; kr[i][3] = t.w * kf;
        const float4 u = *(const float4*)(vp + cb + (size_t)(r0 + i) * VD + c0);
        vr[i][0] = u.x; vr[i][1] = u.y; vr[i][2] = u.z; vr[i][3] = u.w;
      }
#pragma unroll
      for (int i = 0; i < 4; ++i) {
        bf16* d = kb + (r0 + i) * KD + c0;
#pragma unroll
        for (int j = 0; j < 4; ++j) d[j] = (bf16)kr[i][j];
      }
#pragma unroll
      for (int j = 0; j < 4; ++j) {
        bf16* dk = kbT + (c0 + j) * BTC + r0;
        bf16* dv = vsT + (c0 + j) * BTC + r0;
#pragma unroll
        for (int i = 0; i < 4; ++i) { dk[i] = (bf16)kr[i][j]; dv[i] = (bf16)vr[i][j]; }
      }
    }
    if (c + 1 < NT) { // prefetch next chunk q/k/v (global_prefetch_b8)
      __builtin_prefetch(qp + cb + BTC * KD + tid * 32, 0, 0);
      __builtin_prefetch(kp + cb + BTC * KD + tid * 32, 0, 0);
      __builtin_prefetch(vp + cb + BTC * VD + tid * 32, 0, 0);
    }
    // dump entering C state to CbT (transposed, packed 16B stores)
#pragma unroll
    for (int vt = 0; vt < 8; ++vt) {
      v8bf pk;
#pragma unroll
      for (int r = 0; r < 8; ++r) pk[r] = (bf16)cfr[vt][r];
      *(v8bf*)(CbT + (vt * 16 + lx) * KD + wv * 16 + hl * 8) = pk;
    }
    __syncthreads();

    // ---------- S = causal_mask( (qbar @ kbar^T) * rdec ) ----------
    {
      const int t0 = wv * 2;
#pragma unroll
      for (int s = 0; s < 2; ++s) {
        const int iT = (t0 + s) >> 2, jT = (t0 + s) & 3;
        v8f acc = vzero;
        if (jT <= iT) { // tiles strictly above diagonal are fully masked
#pragma unroll
          for (int kk = 0; kk < KD; kk += 32) {
            v16bf a = frag_a(qs, KD, iT * 16, kk, lane);
            v16bf b = frag_bt(kb, KD, kk, jT * 16, lane);
            acc = __builtin_amdgcn_wmma_f32_16x16x32_bf16(false, a, false, b, (short)0, acc, false, false);
          }
        }
        const int j = jT * 16 + lx;
#pragma unroll
        for (int r = 0; r < 8; ++r) {
          const int i = iT * 16 + r + 8 * hl;
          const float sv = (j <= i) ? acc[r] * rdec : 0.0f; // select, no branch
          Sb[i * BTC + j] = (bf16)sv;
        }
      }
    }
    __syncthreads();

    // ---------- denominator: qbar@n + rowsum(S) ----------
    if (tid < BTC) {
      float rs = 0.0f;
#pragma unroll 8
      for (int j = 0; j < BTC; ++j) rs += (float)Sb[tid * BTC + j];
      float qn = 0.0f;
#pragma unroll 8
      for (int kk = 0; kk < KD; ++kk) qn += (float)qs[tid * KD + kk] * sN[kk];
      const float den = qn + rs;
      const float nrm = fmaxf(fabsf(den), exp2f(-sMC[tid]));
      sRn[tid] = 1.0f / nrm;
    }
    __syncthreads();

    // ---------- H = qbar@C + S@V, normalize, store ----------
    {
      const int t0 = wv * 4;
      float* orow = op + (size_t)(c * BTC) * VD;
#pragma unroll
      for (int s = 0; s < 4; ++s) {
        const int iT = (t0 + s) >> 3, vT = (t0 + s) & 7;
        v8f acc = vzero;
#pragma unroll
        for (int kk = 0; kk < KD; kk += 32) {
          v16bf a = frag_a(qs, KD, iT * 16, kk, lane);
          v16bf b = frag_bt(CbT, KD, kk, vT * 16, lane);
          acc = __builtin_amdgcn_wmma_f32_16x16x32_bf16(false, a, false, b, (short)0, acc, false, false);
        }
#pragma unroll
        for (int jj = 0; jj < BTC; jj += 32) {
          v16bf a = frag_a(Sb, BTC, iT * 16, jj, lane);
          v16bf b = frag_bt(vsT, BTC, jj, vT * 16, lane);
          acc = __builtin_amdgcn_wmma_f32_16x16x32_bf16(false, a, false, b, (short)0, acc, false, false);
        }
#pragma unroll
        for (int r = 0; r < 8; ++r) {
          const int i = iT * 16 + r + 8 * hl;
          orow[i * VD + vT * 16 + lx] = acc[r] * sRn[i];
        }
      }
    }

    // ---------- state update: C = C*dec + kbar^T @ V (register accumulators) ----------
#pragma unroll
    for (int vt = 0; vt < 8; ++vt) {
      v8f cc = cfr[vt];
#pragma unroll
      for (int r = 0; r < 8; ++r) cc[r] *= dec;
#pragma unroll
      for (int jj = 0; jj < BTC; jj += 32) {
        v16bf a = frag_a(kbT, BTC, wv * 16, jj, lane);
        v16bf b = frag_bt(vsT, BTC, jj, vt * 16, lane);
        cc = __builtin_amdgcn_wmma_f32_16x16x32_bf16(false, a, false, b, (short)0, cc, false, false);
      }
      cfr[vt] = cc;
    }
    // n, m update
    if (tid < KD) {
      float nn = sN[tid] * dec;
#pragma unroll 8
      for (int j = 0; j < BTC; ++j) nn += (float)kbT[tid * BTC + j];
      sN[tid] = nn;
    }
    if (tid == 0) sM[0] = m_next;
    __syncthreads();
  }
}

extern "C" void kernel_launch(void* const* d_in, const int* in_sizes, int n_in,
                              void* d_out, int out_size, void* d_ws, size_t ws_size,
                              hipStream_t stream) {
  (void)in_sizes; (void)n_in; (void)out_size; (void)d_ws; (void)ws_size;
  const float* q  = (const float*)d_in[0];
  const float* k  = (const float*)d_in[1];
  const float* v  = (const float*)d_in[2];
  const float* ig = (const float*)d_in[3];
  const float* fg = (const float*)d_in[4];
  float* out = (float*)d_out;
  dim3 grid(4 * 8);   // one workgroup per (b,h); chunk recurrence is sequential inside
  dim3 block(256);    // 8 waves (wave32)
  hipLaunchKernelGGL(mlstm_chunk_kernel, grid, block, 0, stream, q, k, v, ig, fg, out);
}
